// MultiHeadAttention_52467320488538
// MI455X (gfx1250) — matrix operand order repaired and checked by reading, not compiled, not run on hardware
//
#include <hip/hip_runtime.h>
#include <hip/hip_bf16.h>

#define BATCH  4
#define SEQ    2048
#define DMODEL 1024
#define NHEAD  16
#define DK     64
#define MROWS  (BATCH * SEQ)   // 8192

typedef __attribute__((ext_vector_type(16))) _Float16 v16h;
typedef __attribute__((ext_vector_type(8)))  float    v8f;

static __device__ __forceinline__ v8f wmma_f16(const v16h& a, const v16h& b, const v8f& c) {
    // V_WMMA_F32_16X16X32_F16 : D = A(16x32) * B(32x16) + C(16x16 f32)
    return __builtin_amdgcn_wmma_f32_16x16x32_f16(
        false, a, false, b, (short)0, c, false, false);
}

// ---- 16-lane reductions with DPP fused directly into the ALU op -----------
// One instruction per step (no NaN-canonicalization, no ds traffic):
// quad_perm xor1, quad_perm xor2, row_half_mirror, row_mirror.
static __device__ __forceinline__ float redmax16(float x) {
    asm("v_max_num_f32_dpp %0, %0, %0 quad_perm:[1,0,3,2] row_mask:0xf bank_mask:0xf" : "+v"(x));
    asm("v_max_num_f32_dpp %0, %0, %0 quad_perm:[2,3,0,1] row_mask:0xf bank_mask:0xf" : "+v"(x));
    asm("v_max_num_f32_dpp %0, %0, %0 row_half_mirror row_mask:0xf bank_mask:0xf"     : "+v"(x));
    asm("v_max_num_f32_dpp %0, %0, %0 row_mirror row_mask:0xf bank_mask:0xf"          : "+v"(x));
    return x;
}
static __device__ __forceinline__ float redsum16(float x) {
    asm("v_add_f32_dpp %0, %0, %0 quad_perm:[1,0,3,2] row_mask:0xf bank_mask:0xf" : "+v"(x));
    asm("v_add_f32_dpp %0, %0, %0 quad_perm:[2,3,0,1] row_mask:0xf bank_mask:0xf" : "+v"(x));
    asm("v_add_f32_dpp %0, %0, %0 row_half_mirror row_mask:0xf bank_mask:0xf"     : "+v"(x));
    asm("v_add_f32_dpp %0, %0, %0 row_mirror row_mask:0xf bank_mask:0xf"          : "+v"(x));
    return x;
}

// ---------------------------------------------------------------------------
// Elementwise f32 -> f16 (weights pre-conversion)
// ---------------------------------------------------------------------------
__global__ __launch_bounds__(256) void cvt_f32_f16_kernel(
    const float* __restrict__ s, _Float16* __restrict__ d, int n)
{
    const int i = (blockIdx.x * 256 + threadIdx.x) * 4;
    if (i + 3 < n) {
        #pragma unroll
        for (int j = 0; j < 4; ++j) d[i + j] = (_Float16)s[i + j];
    }
}

// ---------------------------------------------------------------------------
// GEMM:  C[M x N] = A[M x K] @ W16[N x K]^T + bias[N]
// W tiles double-buffered into LDS with global_load_async_to_lds_b128 and
// shared by all 8 waves. Wave tile 32x64 = 8 WMMAs per k-step.
// ---------------------------------------------------------------------------
template<bool A_HALF, bool OUT_F32>
__global__ __launch_bounds__(256) void gemm_wmma_kernel(
    const void* __restrict__ Aptr, const _Float16* __restrict__ W,
    const float* __restrict__ bias, void* __restrict__ Cptr,
    int N, int K)
{
    const int lane = threadIdx.x & 31;
    const int wave = threadIdx.x >> 5;
    const int ln   = lane & 15;
    const int grp  = lane >> 4;                     // half-wave
    const int mbase = (blockIdx.x * 8 + wave) * 32; // two 16-row tiles
    const int n0    = blockIdx.y * 64;

    // padded row stride 40 halfs (80 B, 16B-aligned; (ln*20)%64 all distinct)
    __shared__ _Float16 lds_w[2][64 * 40];

    // staging role: 256 threads x 16 B  ==  64 n-rows x 32 k halfs
    const int tt = threadIdx.x;
    const int nl = tt >> 2;
    const int c4 = tt & 3;
    const unsigned lds_w0    = (unsigned)(unsigned long long)(&lds_w[0][0]);
    const unsigned stage_off = (unsigned)((nl * 40 + c4 * 8) * 2);
    const _Float16* wsrc = W + (size_t)(n0 + nl) * K + c4 * 8;

    { // prologue: stage k0 = 0 into buffer 0
        unsigned long long ga = (unsigned long long)(const void*)wsrc;
        unsigned la = lds_w0 + stage_off;
        asm volatile("global_load_async_to_lds_b128 %0, %1, off"
                     :: "v"(la), "v"(ga) : "memory");
    }

    v8f acc[2][4] = {};

    for (int k0 = 0; k0 < K; k0 += 32) {
        asm volatile("s_wait_asynccnt 0" ::: "memory");
        __syncthreads();                       // tile k0 visible everywhere
        const int buf = (k0 >> 5) & 1;
        if (k0 + 32 < K) {                     // prefetch next tile (async DMA)
            unsigned long long ga = (unsigned long long)(const void*)(wsrc + k0 + 32);
            unsigned la = lds_w0 + (unsigned)((buf ^ 1) * (64 * 40 * 2)) + stage_off;
            asm volatile("global_load_async_to_lds_b128 %0, %1, off"
                         :: "v"(la), "v"(ga) : "memory");
        }

        // A fragments (16-bit A 16x32 layout), two 16-row tiles
        v16h a[2];
        const int ka = k0 + grp * 8;
        #pragma unroll
        for (int m2 = 0; m2 < 2; ++m2) {
            const int row = mbase + m2 * 16 + ln;
            if (A_HALF) {
                const _Float16* Arow = (const _Float16*)Aptr + (size_t)row * K + ka;
                #pragma unroll
                for (int h = 0; h < 8; ++h) { a[m2][h] = Arow[h]; a[m2][8+h] = Arow[16+h]; }
            } else {
                const float* Arow = (const float*)Aptr + (size_t)row * K + ka;
                #pragma unroll
                for (int h = 0; h < 8; ++h) {
                    a[m2][h]     = (_Float16)Arow[h];
                    a[m2][8 + h] = (_Float16)Arow[16 + h];
                }
            }
        }

        // 4 B fragments from LDS, 8 WMMAs
        const _Float16* wl = &lds_w[buf][0];
        v16h bfr[4];
        #pragma unroll
        for (int t = 0; t < 4; ++t) {
            const _Float16* bp = wl + (t * 16 + ln) * 40 + grp * 8;
            #pragma unroll
            for (int h = 0; h < 8; ++h) { bfr[t][h] = bp[h]; bfr[t][8 + h] = bp[16 + h]; }
        }
        #pragma unroll
        for (int t = 0; t < 4; ++t) {
            acc[0][t] = wmma_f16(a[0], bfr[t], acc[0][t]);
            acc[1][t] = wmma_f16(a[1], bfr[t], acc[1][t]);
        }
    }

    // Epilogue: C/D layout row = r + 8*grp, col = ln
    #pragma unroll
    for (int t = 0; t < 4; ++t) {
        const int ncol = n0 + t * 16 + ln;
        const float bv = bias[ncol];
        #pragma unroll
        for (int m2 = 0; m2 < 2; ++m2) {
            #pragma unroll
            for (int r = 0; r < 8; ++r) {
                const int orow = mbase + m2 * 16 + r + 8 * grp;
                const float val = acc[m2][t][r] + bv;
                if (OUT_F32) ((float*)Cptr)[(size_t)orow * N + ncol] = val;
                else         ((_Float16*)Cptr)[(size_t)orow * N + ncol] = (_Float16)val;
            }
        }
    }
}

// ---------------------------------------------------------------------------
// Flash attention: one (batch,head) per blockIdx.y, 64 q-rows per block
// (4 waves x 16). K tiles async-DMA'd to LDS (double buffered); V tiles
// loaded coalesced and stored transposed [d][key]. Softmax scale folded
// into Q fragments; reductions are single-instruction DPP ops.
// ---------------------------------------------------------------------------
__global__ __launch_bounds__(128) void flash_attn_kernel(
    const _Float16* __restrict__ Qp, const _Float16* __restrict__ Kp,
    const _Float16* __restrict__ Vp, _Float16* __restrict__ Op)
{
    const int lane = threadIdx.x & 31;
    const int wave = threadIdx.x >> 5;
    const int ln   = lane & 15;
    const int grp  = lane >> 4;
    const int bh   = blockIdx.y;
    const int b    = bh >> 4;      // NHEAD == 16
    const int h    = bh & 15;
    const int qrow0 = (blockIdx.x * 4 + wave) * 16;

    __shared__ _Float16 lds_k [2][32 * 72];   // [key][d], stride 72 (conflict-free)
    __shared__ _Float16 lds_vt[2][64 * 40];   // [d][key], stride 40 (conflict-free)
    __shared__ _Float16 lds_p [4][16 * 40];   // per-wave P transpose slice
    _Float16* Pt = lds_p[wave];

    const size_t base = ((size_t)b * SEQ) * DMODEL + (size_t)h * DK;

    // staging roles: 128 threads, 32 B each per tile
    const int tt = threadIdx.x;
    const int kr = tt >> 2;        // key row 0..31
    const int c4 = tt & 3;         // 16-half chunk
    const unsigned ldsk0 = (unsigned)(unsigned long long)(&lds_k[0][0]);
    const unsigned stage_k_off = (unsigned)((kr * 72 + c4 * 16) * 2);
    const _Float16* ksrc = Kp + base + (size_t)kr * DMODEL + c4 * 16;
    const _Float16* vsrc = Vp + base + (size_t)kr * DMODEL + c4 * 16;

    auto stage = [&](int kb, int buf) {
        // K: async DMA global -> LDS (IOFFSET applies to both addresses)
        unsigned long long ga = (unsigned long long)(const void*)(ksrc + (size_t)kb * DMODEL);
        unsigned la = ldsk0 + (unsigned)(buf * (32 * 72 * 2)) + stage_k_off;
        asm volatile("global_load_async_to_lds_b128 %0, %1, off"
                     :: "v"(la), "v"(ga) : "memory");
        asm volatile("global_load_async_to_lds_b128 %0, %1, off offset:16"
                     :: "v"(la), "v"(ga) : "memory");
        // V: coalesced global load, transposed store into [d][key]
        const _Float16* vp2 = vsrc + (size_t)kb * DMODEL;
        _Float16 vv[16];
        #pragma unroll
        for (int i = 0; i < 16; ++i) vv[i] = vp2[i];
        _Float16* vt = &lds_vt[buf][0];
        #pragma unroll
        for (int i = 0; i < 16; ++i) vt[(c4 * 16 + i) * 40 + kr] = vv[i];
    };

    // Resident Q fragments, pre-scaled by 1/sqrt(DK) = 0.125 (exact in f16)
    v16h qa[2];
    {
        const _Float16 qs = (_Float16)0.125f;
        const _Float16* Qrow = Qp + base + (size_t)(qrow0 + ln) * DMODEL;
        #pragma unroll
        for (int ks = 0; ks < 2; ++ks) {
            const int ka = ks * 32 + grp * 8;
            #pragma unroll
            for (int hh = 0; hh < 8; ++hh) {
                qa[ks][hh]     = Qrow[ka + hh]      * qs;
                qa[ks][8 + hh] = Qrow[ka + 16 + hh] * qs;
            }
        }
    }

    v8f o[4] = {};
    float mrow[8], lrow[8];
    #pragma unroll
    for (int r = 0; r < 8; ++r) { mrow[r] = -1e30f; lrow[r] = 0.0f; }

    stage(0, 0);
    for (int kb = 0; kb < SEQ; kb += 32) {
        const int buf = (kb >> 5) & 1;
        asm volatile("s_wait_asynccnt 0" ::: "memory");
        __syncthreads();                       // tile kb ready; prior buf free
        if (kb + 32 < SEQ) stage(kb + 32, buf ^ 1);   // overlap next-tile fetch

        // S = Q @ K^T (16x32): 4 B-frags, 4 WMMAs
        const _Float16* kl = &lds_k[buf][0];
        v16h kfr[2][2];
        #pragma unroll
        for (int nt = 0; nt < 2; ++nt) {
            #pragma unroll
            for (int ks = 0; ks < 2; ++ks) {
                const _Float16* bp = kl + (nt * 16 + ln) * 72 + ks * 32 + grp * 8;
                #pragma unroll
                for (int hh = 0; hh < 8; ++hh) { kfr[nt][ks][hh] = bp[hh]; kfr[nt][ks][8 + hh] = bp[16 + hh]; }
            }
        }
        v8f sacc[2] = {};
        #pragma unroll
        for (int nt = 0; nt < 2; ++nt) {
            sacc[nt] = wmma_f16(qa[0], kfr[nt][0], sacc[nt]);
            sacc[nt] = wmma_f16(qa[1], kfr[nt][1], sacc[nt]);
        }

        // ---- online softmax (scores pre-scaled; fused-DPP reductions) ----
        float sf[8], psum[8];
        #pragma unroll
        for (int r = 0; r < 8; ++r) {
            const float pm = redmax16(fmaxf(sacc[0][r], sacc[1][r]));
            const float mnew = fmaxf(mrow[r], pm);
            sf[r]   = __expf(mrow[r] - mnew);
            mrow[r] = mnew;
        }
        #pragma unroll
        for (int nt = 0; nt < 2; ++nt) {
            #pragma unroll
            for (int r = 0; r < 8; ++r) {
                const float e = __expf(sacc[nt][r] - mrow[r]);
                if (nt == 0) psum[r] = e; else psum[r] += e;
                Pt[(r + 8 * grp) * 40 + nt * 16 + ln] = (_Float16)e;
            }
        }
        #pragma unroll
        for (int r = 0; r < 8; ++r) lrow[r] = lrow[r] * sf[r] + redsum16(psum[r]);
        #pragma unroll
        for (int t = 0; t < 4; ++t) {
            #pragma unroll
            for (int r = 0; r < 8; ++r) o[t][r] *= sf[r];
        }

        // intra-wave LDS RAW on P before transposed read-back
        asm volatile("s_wait_dscnt 0" ::: "memory");
        v16h pfrag;    // P as 16x32 A-fragment
        {
            const _Float16* prow = Pt + ln * 40 + grp * 8;
            #pragma unroll
            for (int hh = 0; hh < 8; ++hh) {
                pfrag[hh]     = prow[hh];
                pfrag[8 + hh] = prow[16 + hh];
            }
        }
        // O += P(16x32) @ V(32x64): 4 B-frags, 4 WMMAs
        const _Float16* vl = &lds_vt[buf][0];
        v16h vfr[4];
        #pragma unroll
        for (int t = 0; t < 4; ++t) {
            const _Float16* bp = vl + (t * 16 + ln) * 40 + grp * 8;
            #pragma unroll
            for (int hh = 0; hh < 8; ++hh) { vfr[t][hh] = bp[hh]; vfr[t][8 + hh] = bp[16 + hh]; }
        }
        #pragma unroll
        for (int t = 0; t < 4; ++t) o[t] = wmma_f16(pfrag, vfr[t], o[t]);
    }

    // normalize (one reciprocal per row), emit f16 attention output
    float inv[8];
    #pragma unroll
    for (int r = 0; r < 8; ++r) inv[r] = __frcp_rn(lrow[r]);
    #pragma unroll
    for (int t = 0; t < 4; ++t) {
        #pragma unroll
        for (int r = 0; r < 8; ++r) {
            const int orow = qrow0 + r + 8 * grp;
            Op[base + (size_t)orow * DMODEL + t * 16 + ln] =
                (_Float16)(o[t][r] * inv[r]);
        }
    }
}

// ---------------------------------------------------------------------------
extern "C" void kernel_launch(void* const* d_in, const int* in_sizes, int n_in,
                              void* d_out, int out_size, void* d_ws, size_t ws_size,
                              hipStream_t stream) {
    (void)in_sizes; (void)n_in; (void)out_size; (void)ws_size;
    const float* q  = (const float*)d_in[0];
    const float* k  = (const float*)d_in[1];
    const float* v  = (const float*)d_in[2];
    const float* Wq = (const float*)d_in[3];
    const float* bq = (const float*)d_in[4];
    const float* Wk = (const float*)d_in[5];
    const float* bk = (const float*)d_in[6];
    const float* Wv = (const float*)d_in[7];
    const float* bv = (const float*)d_in[8];
    const float* Wo = (const float*)d_in[9];
    const float* bo = (const float*)d_in[10];
    float* out = (float*)d_out;

    const size_t act_bytes = (size_t)MROWS * DMODEL * sizeof(_Float16);   // 16 MB
    const size_t w_bytes   = (size_t)DMODEL * DMODEL * sizeof(_Float16);  //  2 MB
    char* ws = (char*)d_ws;
    _Float16* Qp   = (_Float16*)(ws + 0 * act_bytes);
    _Float16* Kp   = (_Float16*)(ws + 1 * act_bytes);
    _Float16* Vp   = (_Float16*)(ws + 2 * act_bytes);
    _Float16* Ap   = (_Float16*)(ws + 3 * act_bytes);
    _Float16* W16q = (_Float16*)(ws + 4 * act_bytes + 0 * w_bytes);
    _Float16* W16k = (_Float16*)(ws + 4 * act_bytes + 1 * w_bytes);
    _Float16* W16v = (_Float16*)(ws + 4 * act_bytes + 2 * w_bytes);
    _Float16* W16o = (_Float16*)(ws + 4 * act_bytes + 3 * w_bytes);      // 72 MB total

    const int wn = DMODEL * DMODEL;
    dim3 cblk(256), cgrid(wn / 1024);
    cvt_f32_f16_kernel<<<cgrid, cblk, 0, stream>>>(Wq, W16q, wn);
    cvt_f32_f16_kernel<<<cgrid, cblk, 0, stream>>>(Wk, W16k, wn);
    cvt_f32_f16_kernel<<<cgrid, cblk, 0, stream>>>(Wv, W16v, wn);
    cvt_f32_f16_kernel<<<cgrid, cblk, 0, stream>>>(Wo, W16o, wn);

    dim3 gblk(256), ggrid(MROWS / 256, DMODEL / 64);   // 32 x 16 blocks
    gemm_wmma_kernel<false, false><<<ggrid, gblk, 0, stream>>>(q, W16q, bq, Qp, DMODEL, DMODEL);
    gemm_wmma_kernel<false, false><<<ggrid, gblk, 0, stream>>>(k, W16k, bk, Kp, DMODEL, DMODEL);
    gemm_wmma_kernel<false, false><<<ggrid, gblk, 0, stream>>>(v, W16v, bv, Vp, DMODEL, DMODEL);

    dim3 ablk(128), agrid(SEQ / 64, BATCH * NHEAD);    // 32 x 64 blocks
    flash_attn_kernel<<<agrid, ablk, 0, stream>>>(Qp, Kp, Vp, Ap);

    gemm_wmma_kernel<true, true><<<ggrid, gblk, 0, stream>>>(Ap, W16o, bo, out, DMODEL, DMODEL);
}
